// MHN_46377056862344
// MI455X (gfx1250) — compile-verified, matrix operand order checked by reading, not run on hardware
//
#include <hip/hip_runtime.h>

// ---------------------------------------------------------------------------
// Types for CDNA5 WMMA (wave32): v_wmma_f32_16x16x32_bf16
// ---------------------------------------------------------------------------
typedef __attribute__((ext_vector_type(16))) __bf16 bf16x16;
typedef __attribute__((ext_vector_type(4)))  __bf16 bf16x4;
typedef __attribute__((ext_vector_type(8)))  float  f32x8;
typedef __attribute__((vector_size(4 * sizeof(int)))) int v4i;

#define N_TEMP 100000
#define BS     512
#define DD     1024
#define BETA   0.125f
#define LN_EPS 1e-5f

#if defined(__has_builtin)
#if __has_builtin(__builtin_amdgcn_global_load_async_to_lds_b128)
#define HAVE_ASYNC_LDS 1
#endif
#endif

// ---------------------------------------------------------------------------
// fp32 -> bf16 conversion (for weight matrices)
// ---------------------------------------------------------------------------
__global__ __launch_bounds__(256)
void cvt_f32_bf16_kernel(const float* __restrict__ src, __bf16* __restrict__ dst, int n4) {
    int i = blockIdx.x * 256 + threadIdx.x;
    if (i < n4) {
        float4 v = ((const float4*)src)[i];
        bf16x4 o;
        o.x = (__bf16)v.x; o.y = (__bf16)v.y; o.z = (__bf16)v.z; o.w = (__bf16)v.w;
        ((bf16x4*)dst)[i] = o;
    }
}

// ---------------------------------------------------------------------------
// LayerNorm (elementwise_affine=False) over D=1024, fp32 in -> bf16 out.
// ---------------------------------------------------------------------------
__global__ __launch_bounds__(256)
void ln_in_kernel(const float* __restrict__ x, __bf16* __restrict__ out) {
    const int row = blockIdx.x;
    const int tid = threadIdx.x;
    const float4 v = ((const float4*)(x + (size_t)row * DD))[tid];

    __shared__ float red[256];
    red[tid] = v.x + v.y + v.z + v.w;
    __syncthreads();
    #pragma unroll
    for (int off = 128; off > 0; off >>= 1) {
        if (tid < off) red[tid] += red[tid + off];
        __syncthreads();
    }
    const float mu = red[0] * (1.0f / DD);
    __syncthreads();

    const float d0 = v.x - mu, d1 = v.y - mu, d2 = v.z - mu, d3 = v.w - mu;
    red[tid] = d0 * d0 + d1 * d1 + d2 * d2 + d3 * d3;
    __syncthreads();
    #pragma unroll
    for (int off = 128; off > 0; off >>= 1) {
        if (tid < off) red[tid] += red[tid + off];
        __syncthreads();
    }
    const float rinv = rsqrtf(red[0] * (1.0f / DD) + LN_EPS);

    bf16x4 o;
    o.x = (__bf16)(d0 * rinv); o.y = (__bf16)(d1 * rinv);
    o.z = (__bf16)(d2 * rinv); o.w = (__bf16)(d3 * rinv);
    ((bf16x4*)(out + (size_t)row * DD))[tid] = o;
}

// ---------------------------------------------------------------------------
// LayerNorm over bf16 H + fp32 residual -> bf16 X.   X = LN(H) + residual
// ---------------------------------------------------------------------------
__global__ __launch_bounds__(256)
void ln_res_kernel(const __bf16* __restrict__ h, const float* __restrict__ resid,
                   __bf16* __restrict__ out) {
    const int row = blockIdx.x;
    const int tid = threadIdx.x;
    const bf16x4 hv = ((const bf16x4*)(h + (size_t)row * DD))[tid];
    const float4 rv = ((const float4*)(resid + (size_t)row * DD))[tid];
    const float h0 = (float)hv.x, h1 = (float)hv.y, h2 = (float)hv.z, h3 = (float)hv.w;

    __shared__ float red[256];
    red[tid] = h0 + h1 + h2 + h3;
    __syncthreads();
    #pragma unroll
    for (int off = 128; off > 0; off >>= 1) {
        if (tid < off) red[tid] += red[tid + off];
        __syncthreads();
    }
    const float mu = red[0] * (1.0f / DD);
    __syncthreads();

    const float d0 = h0 - mu, d1 = h1 - mu, d2 = h2 - mu, d3 = h3 - mu;
    red[tid] = d0 * d0 + d1 * d1 + d2 * d2 + d3 * d3;
    __syncthreads();
    #pragma unroll
    for (int off = 128; off > 0; off >>= 1) {
        if (tid < off) red[tid] += red[tid + off];
        __syncthreads();
    }
    const float rinv = rsqrtf(red[0] * (1.0f / DD) + LN_EPS);

    bf16x4 o;
    o.x = (__bf16)(d0 * rinv + rv.x); o.y = (__bf16)(d1 * rinv + rv.y);
    o.z = (__bf16)(d2 * rinv + rv.z); o.w = (__bf16)(d3 * rinv + rv.w);
    ((bf16x4*)(out + (size_t)row * DD))[tid] = o;
}

// ---------------------------------------------------------------------------
// Tiled bf16 WMMA GEMM:  C[M,N] = A[M,K] * B[N,K]^T   (K contiguous in both)
// MODE 0: C bf16 = acc + bias[n]          (encoder linear)
// MODE 1: C fp32 = acc * scale            (beta-scaled attention scores)
// Block = 256 threads = 8 waves (wave32). Tile BM=128, BN=128, BK=32,
// double-buffered LDS; async global->LDS staging (ASYNCcnt) when available.
// Wave grid 2x4; each wave computes a 64x32 patch = 8 f32x8 accumulators,
// 8 WMMAs per K-step. OOB rows are clamped (their C entries are never stored),
// so the hot loop has no exec-mask predication.
// ---------------------------------------------------------------------------
template <int MODE>
__global__ __launch_bounds__(256)
void gemm_bf16_kernel(const __bf16* __restrict__ A, const __bf16* __restrict__ B,
                      const float* __restrict__ bias, void* __restrict__ Cout,
                      int M, int N, int K, float scale) {
    constexpr int BM = 128, BN = 128, BK = 32;
    __shared__ alignas(64) __bf16 As[2][BM * BK];
    __shared__ alignas(64) __bf16 Bs[2][BN * BK];

    const int tid  = threadIdx.x;
    const int lane = tid & 31;
    const int wave = tid >> 5;
    const int mW   = wave >> 2;   // 0..1  -> 64-row slab
    const int nW   = wave & 3;    // 0..3  -> 32-col slab

    const int blockM = blockIdx.y * BM;
    const int blockN = blockIdx.x * BN;

    // Loaders: 128 rows x 4 chunks(16B) = 512 slots per matrix, 2 per thread.
    int lrow[2], lko[2], rowA[2], rowB[2];
    #pragma unroll
    for (int j = 0; j < 2; ++j) {
        const int chunk = tid + j * 256;
        lrow[j] = chunk >> 2;
        lko[j]  = (chunk & 3) << 3;   // bf16 element offset 0,8,16,24
        int ra = blockM + lrow[j]; if (ra > M - 1) ra = M - 1;
        int rb = blockN + lrow[j]; if (rb > N - 1) rb = N - 1;
        rowA[j] = ra; rowB[j] = rb;
    }

    const f32x8 zero = {0.f, 0.f, 0.f, 0.f, 0.f, 0.f, 0.f, 0.f};
    f32x8 c[4][2];
    #pragma unroll
    for (int mi = 0; mi < 4; ++mi) { c[mi][0] = zero; c[mi][1] = zero; }

    const int lr = lane & 15;          // row within 16-tile
    const int lk = (lane >> 4) << 4;   // K half select (0 / 16)

#define GEMM_STAGE(K0, S)                                                        \
    do {                                                                         \
        _Pragma("unroll")                                                        \
        for (int j = 0; j < 2; ++j) {                                            \
            STAGE_ONE(A + (size_t)rowA[j] * K + (K0) + lko[j],                   \
                      &As[(S)][lrow[j] * BK + lko[j]]);                          \
            STAGE_ONE(B + (size_t)rowB[j] * K + (K0) + lko[j],                   \
                      &Bs[(S)][lrow[j] * BK + lko[j]]);                          \
        }                                                                        \
    } while (0)

#if defined(HAVE_ASYNC_LDS)
#define STAGE_ONE(GP, LP)                                                        \
    __builtin_amdgcn_global_load_async_to_lds_b128(                              \
        (v4i*)(__bf16*)(GP), (v4i*)(LP), 0, 0)
#if __has_builtin(__builtin_amdgcn_s_wait_asynccnt)
#define GEMM_STAGE_WAIT()                                                        \
    do { __builtin_amdgcn_s_wait_asynccnt(0); } while (0)
#else
#define GEMM_STAGE_WAIT()                                                        \
    do { asm volatile("s_wait_asynccnt 0x0" ::: "memory"); } while (0)
#endif
#else
#define STAGE_ONE(GP, LP)                                                        \
    do { *(uint4*)(LP) = *(const uint4*)(GP); } while (0)
#define GEMM_STAGE_WAIT() do { } while (0)
#endif

    // Prologue: stage K-slab 0 into buffer 0.
    GEMM_STAGE(0, 0);
    GEMM_STAGE_WAIT();
    __syncthreads();

    const int nIter = K / BK;
    for (int it = 0; it < nIter; ++it) {
        const int cur = it & 1;
        if (it + 1 < nIter) GEMM_STAGE((it + 1) * BK, cur ^ 1);

        bf16x16 afrag[4], bfrag[2];
        #pragma unroll
        for (int mi = 0; mi < 4; ++mi)
            afrag[mi] = *(const bf16x16*)&As[cur][(mW * 64 + mi * 16 + lr) * BK + lk];
        #pragma unroll
        for (int ni = 0; ni < 2; ++ni)
            bfrag[ni] = *(const bf16x16*)&Bs[cur][(nW * 32 + ni * 16 + lr) * BK + lk];

        #pragma unroll
        for (int mi = 0; mi < 4; ++mi)
            #pragma unroll
            for (int ni = 0; ni < 2; ++ni)
                c[mi][ni] = __builtin_amdgcn_wmma_f32_16x16x32_bf16(
                    false, afrag[mi], false, bfrag[ni], (short)0, c[mi][ni],
                    false, false);

        GEMM_STAGE_WAIT();
        __syncthreads();
    }

#undef GEMM_STAGE
#undef STAGE_ONE
#undef GEMM_STAGE_WAIT

    // Epilogue. C/D layout: VGPR r, lanes 0-15 -> (M=r, N=lane);
    //                        lanes 16-31 -> (M=r+8, N=lane-16)
    const int rowAdj = (lane >= 16) ? 8 : 0;
    const int colLoc = lane & 15;
    #pragma unroll
    for (int mi = 0; mi < 4; ++mi) {
        #pragma unroll
        for (int ni = 0; ni < 2; ++ni) {
            #pragma unroll
            for (int r = 0; r < 8; ++r) {
                const int row = blockM + mW * 64 + mi * 16 + r + rowAdj;
                const int col = blockN + nW * 32 + ni * 16 + colLoc;
                if (row < M && col < N) {
                    const float v = c[mi][ni][r];
                    if constexpr (MODE == 0) {
                        ((__bf16*)Cout)[(size_t)row * N + col] = (__bf16)(v + bias[col]);
                    } else {
                        ((float*)Cout)[(size_t)row * N + col] = v * scale;
                    }
                }
            }
        }
    }
}

// ---------------------------------------------------------------------------
// Row softmax in-place over `cols` (cols % 4 == 0). One workgroup per row.
// ---------------------------------------------------------------------------
__global__ __launch_bounds__(256)
void softmax_row_kernel(float* __restrict__ data, int cols) {
    const int row = blockIdx.x;
    const int tid = threadIdx.x;
    float* p = data + (size_t)row * cols;
    const int nv = cols >> 2;

    __shared__ float red[256];

    float m = -3.4028235e38f;
    for (int i = tid; i < nv; i += 256) {
        float4 v = ((const float4*)p)[i];
        m = fmaxf(m, fmaxf(fmaxf(v.x, v.y), fmaxf(v.z, v.w)));
    }
    red[tid] = m;
    __syncthreads();
    #pragma unroll
    for (int off = 128; off > 0; off >>= 1) {
        if (tid < off) red[tid] = fmaxf(red[tid], red[tid + off]);
        __syncthreads();
    }
    m = red[0];
    __syncthreads();

    float s = 0.f;
    for (int i = tid; i < nv; i += 256) {
        float4 v = ((const float4*)p)[i];
        s += __expf(v.x - m) + __expf(v.y - m) + __expf(v.z - m) + __expf(v.w - m);
    }
    red[tid] = s;
    __syncthreads();
    #pragma unroll
    for (int off = 128; off > 0; off >>= 1) {
        if (tid < off) red[tid] += red[tid + off];
        __syncthreads();
    }
    const float inv = 1.0f / red[0];

    for (int i = tid; i < nv; i += 256) {
        float4 v = ((const float4*)p)[i];
        v.x = __expf(v.x - m) * inv; v.y = __expf(v.y - m) * inv;
        v.z = __expf(v.z - m) * inv; v.w = __expf(v.w - m) * inv;
        ((float4*)p)[i] = v;
    }
}

// ---------------------------------------------------------------------------
// Launch: W->bf16; encode templates; encode queries; score GEMM; softmax.
// ---------------------------------------------------------------------------
extern "C" void kernel_launch(void* const* d_in, const int* in_sizes, int n_in,
                              void* d_out, int out_size, void* d_ws, size_t ws_size,
                              hipStream_t stream) {
    const float* m         = (const float*)d_in[0];
    const float* templates = (const float*)d_in[1];
    const float* W_mol     = (const float*)d_in[2];
    const float* b_mol     = (const float*)d_in[3];
    const float* W_temp    = (const float*)d_in[4];
    const float* b_temp    = (const float*)d_in[5];
    float* out = (float*)d_out;

    char* ws = (char*)d_ws;
    size_t off = 0;
    auto alloc = [&](size_t bytes) -> void* {
        void* p = ws + off;
        off = (off + bytes + 255) & ~(size_t)255;
        return p;
    };
    __bf16* bufA = (__bf16*)alloc((size_t)N_TEMP * DD * 2); // LN(templates), then X
    __bf16* bufH = (__bf16*)alloc((size_t)N_TEMP * DD * 2); // H (templates), then Hq
    __bf16* Wt   = (__bf16*)alloc((size_t)DD * DD * 2);
    __bf16* Wm   = (__bf16*)alloc((size_t)DD * DD * 2);
    __bf16* Amol = (__bf16*)alloc((size_t)BS * DD * 2);
    __bf16* Xi   = (__bf16*)alloc((size_t)BS * DD * 2);

    // Weights -> bf16
    const int wn4 = DD * DD / 4;
    cvt_f32_bf16_kernel<<<(wn4 + 255) / 256, 256, 0, stream>>>(W_temp, Wt, wn4);
    cvt_f32_bf16_kernel<<<(wn4 + 255) / 256, 256, 0, stream>>>(W_mol,  Wm, wn4);

    // Template encoder: X = LN(LN(t) @ Wt^T + b) + t
    ln_in_kernel<<<N_TEMP, 256, 0, stream>>>(templates, bufA);
    {
        dim3 grid(DD / 128, (N_TEMP + 127) / 128);
        gemm_bf16_kernel<0><<<grid, 256, 0, stream>>>(bufA, Wt, b_temp, bufH,
                                                      N_TEMP, DD, DD, 1.0f);
    }
    ln_res_kernel<<<N_TEMP, 256, 0, stream>>>(bufH, templates, bufA); // X in bufA

    // Query encoder: Xi = LN(LN(m) @ Wm^T + b) + m
    ln_in_kernel<<<BS, 256, 0, stream>>>(m, Amol);
    {
        dim3 grid(DD / 128, BS / 128);
        gemm_bf16_kernel<0><<<grid, 256, 0, stream>>>(Amol, Wm, b_mol, bufH,
                                                      BS, DD, DD, 1.0f);
    }
    ln_res_kernel<<<BS, 256, 0, stream>>>(bufH, m, Xi);

    // Scores: out = beta * Xi @ X^T, then row softmax
    {
        dim3 grid((N_TEMP + 127) / 128, BS / 128);
        gemm_bf16_kernel<1><<<grid, 256, 0, stream>>>(Xi, bufA, nullptr, out,
                                                      BS, N_TEMP, DD, BETA);
    }
    softmax_row_kernel<<<BS, 256, 0, stream>>>(out, N_TEMP);
}
